// QuantGRU_4973572129348
// MI455X (gfx1250) — 8-bit matrix operand order repaired and checked by reading, compiled, not run on hardware
//
#include <hip/hip_runtime.h>
#include <stdint.h>


typedef __attribute__((ext_vector_type(8))) int v8i;

#define TT   1024
#define BB   64
#define HH   256
#define G3H  768
#define NROW 65536
#define NE   16384

#define WS_SLOTS 0
#define WS_BX    64
#define WS_BR    4096
#define WS_WQT   8192
#define WS_RQT   204800
#define WS_XQ    401408
#define WS_WX    17178624
#define SCAN_LDS 268544

__device__ __forceinline__ int koff_a(int j, int hi) {
  return ((j & 1) << 2) | ((j & 2) << 2) | ((j & 4) << 3) | (hi << 4);
}

__device__ __forceinline__ float qclampf(float q) {
  return fminf(fmaxf(q, -128.f), 127.f);
}
__device__ __forceinline__ float qdq(float v, float s) {
  return qclampf(rintf(v / s)) * s;
}
__device__ __forceinline__ float slot_scale(const unsigned* slot) {
  return fmaxf(__uint_as_float(*slot) * (1.f / 127.f), 1e-8f);
}
__device__ __forceinline__ float wave_max32(float v) {
#pragma unroll
  for (int m = 16; m >= 1; m >>= 1) v = fmaxf(v, __shfl_xor(v, m, 32));
  return v;
}

__global__ void k_init_slots(unsigned* slots) {
  if (threadIdx.x < 8) slots[threadIdx.x] = 0u;
}

__global__ void k_absmax(const float* __restrict__ p, int n, unsigned* slot) {
  float lm = 0.f;
  for (int i = blockIdx.x * blockDim.x + threadIdx.x; i < n;
       i += gridDim.x * blockDim.x)
    lm = fmaxf(lm, fabsf(p[i]));
  lm = wave_max32(lm);
  if ((threadIdx.x & 31) == 0) atomicMax(slot, __float_as_uint(lm));
}

__global__ void k_quant_bias(const float* __restrict__ bih,
                             const float* __restrict__ bhh,
                             float* __restrict__ bx, float* __restrict__ br) {
  __shared__ float red[64];
  const int tid = threadIdx.x;
  const int pt = tid < 256 ? tid + 256 : (tid < 512 ? tid - 256 : tid);
  const float vih = bih[pt], vhh = bhh[pt];

  float m = wave_max32(fabsf(vih));
  if ((tid & 31) == 0) red[tid >> 5] = m;
  __syncthreads();
  if (tid == 0) {
    float g = 0.f;
    for (int i = 0; i < 24; ++i) g = fmaxf(g, red[i]);
    red[32] = g;
  }
  __syncthreads();
  const float sih = fmaxf(red[32] * (1.f / 127.f), 1e-8f);
  __syncthreads();

  m = wave_max32(fabsf(vhh));
  if ((tid & 31) == 0) red[tid >> 5] = m;
  __syncthreads();
  if (tid == 0) {
    float g = 0.f;
    for (int i = 0; i < 24; ++i) g = fmaxf(g, red[i]);
    red[33] = g;
  }
  __syncthreads();
  const float shh = fmaxf(red[33] * (1.f / 127.f), 1e-8f);

  bx[tid] = qdq(vih, sih);
  br[tid] = qdq(vhh, shh);
}

__global__ void k_quant_w(const float* __restrict__ w, const unsigned* slot,
                          signed char* __restrict__ outT) {
  const int idx = blockIdx.x * 256 + threadIdx.x;
  const int n = idx >> 8, k = idx & 255;
  const int pt = n < 256 ? n + 256 : (n < 512 ? n - 256 : n);
  const float s = slot_scale(slot);
  outT[idx] = (signed char)(int)qclampf(rintf(w[pt * 256 + k] / s));
}

__global__ void k_quant_x(const float* __restrict__ x, const unsigned* slot,
                          signed char* __restrict__ xq) {
  const int i4 = blockIdx.x * 256 + threadIdx.x;
  const float s = slot_scale(slot);
  const float4 v = ((const float4*)x)[i4];
  const int q0 = (int)qclampf(rintf(v.x / s));
  const int q1 = (int)qclampf(rintf(v.y / s));
  const int q2 = (int)qclampf(rintf(v.z / s));
  const int q3 = (int)qclampf(rintf(v.w / s));
  ((unsigned*)xq)[i4] = (unsigned)(q0 & 255) | ((unsigned)(q1 & 255) << 8) |
                        ((unsigned)(q2 & 255) << 16) | ((unsigned)(q3 & 255) << 24);
}

__global__ void __launch_bounds__(256)
k_wx_gemm(const signed char* __restrict__ xq, const signed char* __restrict__ WqT,
          unsigned* __restrict__ slots, float* __restrict__ Wx) {
  const int tid = threadIdx.x;
  const int wave = tid >> 5, lane = tid & 31, l15 = lane & 15, hi = lane >> 4;
  const int m0 = blockIdx.x * 32 + (wave & 1) * 16;
  const int n0 = blockIdx.y * 64 + (wave >> 1) * 16;

  const signed char* arow = xq + (size_t)(m0 + l15) * 256;
  const signed char* brow = WqT + (size_t)(n0 + l15) * 256;

  v8i acc;
#pragma unroll
  for (int j = 0; j < 8; ++j) acc[j] = 0;

#pragma unroll
  for (int kc = 0; kc < 4; ++kc) {
    v8i A, B;
#pragma unroll
    for (int j = 0; j < 8; ++j)
      A[j] = *(const int*)(arow + kc * 64 + koff_a(j, hi));
    const uint4 b0 = *(const uint4*)(brow + kc * 64 + hi * 16);
    const uint4 b1 = *(const uint4*)(brow + kc * 64 + 32 + hi * 16);
    B[0] = (int)b0.x; B[1] = (int)b0.y; B[2] = (int)b0.z; B[3] = (int)b0.w;
    B[4] = (int)b1.x; B[5] = (int)b1.y; B[6] = (int)b1.z; B[7] = (int)b1.w;
    acc = __builtin_amdgcn_wmma_i32_16x16x64_iu8(true, A, true, B, acc, false, false);
  }

  const float f = slot_scale(&slots[0]) * slot_scale(&slots[1]);
  float lm = 0.f;
#pragma unroll
  for (int v = 0; v < 8; ++v) {
    const int m = m0 + v + hi * 8;
    const float val = (float)acc[v] * f;
    Wx[(size_t)m * G3H + n0 + l15] = val;
    lm = fmaxf(lm, fabsf(val));
  }
  lm = wave_max32(lm);
  if (lane == 0) atomicMax(&slots[3], __float_as_uint(lm));
}

__device__ __forceinline__ float block_absmax(float v, float* red) {
  v = wave_max32(v);
  const int wv = threadIdx.x >> 5, ln = threadIdx.x & 31;
  if (ln == 0) red[wv] = v;
  __syncthreads();
  if (wv == 0) {
    float w = wave_max32(red[ln]);
    if (ln == 0) red[32] = w;
  }
  __syncthreads();
  return red[32];
}

__device__ __forceinline__ void fq16(float* a, float* red) {
  float lm = 0.f;
#pragma unroll
  for (int c = 0; c < 16; ++c) lm = fmaxf(lm, fabsf(a[c]));
  const float s = fmaxf(block_absmax(lm, red) * (1.f / 127.f), 1e-8f);
#pragma unroll
  for (int c = 0; c < 16; ++c) a[c] = qclampf(rintf(a[c] / s)) * s;
}

__global__ void __launch_bounds__(1024)
k_scan(const float* __restrict__ Wx, const signed char* __restrict__ RqTg,
       const float* __restrict__ bxg, const float* __restrict__ brg,
       const unsigned* __restrict__ slots, const float* __restrict__ h0,
       float* __restrict__ out) {
  extern __shared__ __align__(16) char smem[];
  signed char* RqT = (signed char*)smem;
  signed char* RhQ = (signed char*)(smem + 196608);
  signed char* hq  = (signed char*)(smem + 245760);
  float* bx  = (float*)(smem + 262144);
  float* br  = (float*)(smem + 265216);
  float* red = (float*)(smem + 268288);

  const int tid = threadIdx.x;
  const int wave = tid >> 5, lane = tid & 31, l15 = lane & 15, hi = lane >> 4;
  const int mt = wave & 3, ng = wave >> 2;
  const int m0 = mt * 16;

  for (int i = tid; i < 12288; i += 1024)
    ((uint4*)RqT)[i] = ((const uint4*)RqTg)[i];
  if (tid < G3H) { bx[tid] = bxg[tid]; br[tid] = brg[tid]; }

  float h[16];
#pragma unroll
  for (int c = 0; c < 16; ++c) h[c] = h0[c * 1024 + tid];

  const float sR  = slot_scale(&slots[2]);
  const float sWx = slot_scale(&slots[3]);
  __syncthreads();

  for (int t = 0; t < TT; ++t) {
    float lm = 0.f;
#pragma unroll
    for (int c = 0; c < 16; ++c) lm = fmaxf(lm, fabsf(h[c]));
    const float sH = fmaxf(block_absmax(lm, red) * (1.f / 127.f), 1e-8f);
#pragma unroll
    for (int c = 0; c < 16; ++c)
      hq[c * 1024 + tid] = (signed char)(int)qclampf(rintf(h[c] / sH));
    __syncthreads();

    uint32_t areg[4][8];
    const signed char* arow = hq + (m0 + l15) * 256;
#pragma unroll
    for (int kc = 0; kc < 4; ++kc)
#pragma unroll
      for (int j = 0; j < 8; ++j)
        areg[kc][j] = *(const uint32_t*)(arow + kc * 64 + koff_a(j, hi));

    int accv[6][8];
    int imax = 0;
#pragma unroll
    for (int nt = 0; nt < 6; ++nt) {
      const int n0 = (ng * 6 + nt) * 16;
      const signed char* brow = RqT + (n0 + l15) * 256;
      v8i acc;
#pragma unroll
      for (int j = 0; j < 8; ++j) acc[j] = 0;
#pragma unroll
      for (int kc = 0; kc < 4; ++kc) {
        v8i A, B;
#pragma unroll
        for (int j = 0; j < 8; ++j) A[j] = (int)areg[kc][j];
        const uint4 b0 = *(const uint4*)(brow + kc * 64 + hi * 16);
        const uint4 b1 = *(const uint4*)(brow + kc * 64 + 32 + hi * 16);
        B[0] = (int)b0.x; B[1] = (int)b0.y; B[2] = (int)b0.z; B[3] = (int)b0.w;
        B[4] = (int)b1.x; B[5] = (int)b1.y; B[6] = (int)b1.z; B[7] = (int)b1.w;
        acc = __builtin_amdgcn_wmma_i32_16x16x64_iu8(true, A, true, B, acc,
                                                     false, false);
      }
#pragma unroll
      for (int v = 0; v < 8; ++v) {
        accv[nt][v] = acc[v];
        imax = max(imax, abs(acc[v]));
      }
    }

    const float sHR = sH * sR;
    const float sRh = fmaxf(block_absmax((float)imax * sHR, red) * (1.f / 127.f),
                            1e-8f);
    const float fr = sHR / sRh;
#pragma unroll
    for (int nt = 0; nt < 6; ++nt) {
      const int n0 = (ng * 6 + nt) * 16;
#pragma unroll
      for (int v = 0; v < 8; ++v)
        RhQ[(m0 + v + hi * 8) * G3H + n0 + l15] =
            (signed char)(int)qclampf(rintf((float)accv[nt][v] * fr));
    }
    __syncthreads();

    if (t + 1 < TT)
      __builtin_prefetch(Wx + (size_t)(t + 1) * (BB * G3H) + tid * 48, 0, 0);

    const float* wxt = Wx + (size_t)t * (BB * G3H);
    float z[16], r[16], g[16];

#pragma unroll
    for (int c = 0; c < 16; ++c) {
      const int e = c * 1024 + tid, b = e >> 8, j = e & 255;
      z[c] = qdq(wxt[b * G3H + j], sWx) + bx[j] +
             (float)RhQ[b * G3H + j] * sRh + br[j];
    }
    fq16(z, red);
#pragma unroll
    for (int c = 0; c < 16; ++c) z[c] = 1.f / (1.f + expf(-z[c]));
    fq16(z, red);

#pragma unroll
    for (int c = 0; c < 16; ++c) {
      const int e = c * 1024 + tid, b = e >> 8, j = e & 255;
      r[c] = qdq(wxt[b * G3H + 256 + j], sWx) + bx[256 + j] +
             (float)RhQ[b * G3H + 256 + j] * sRh + br[256 + j];
    }
    fq16(r, red);
#pragma unroll
    for (int c = 0; c < 16; ++c) r[c] = 1.f / (1.f + expf(-r[c]));
    fq16(r, red);

#pragma unroll
    for (int c = 0; c < 16; ++c) {
      const int e = c * 1024 + tid, b = e >> 8, j = e & 255;
      g[c] = (float)RhQ[b * G3H + 512 + j] * sRh + br[512 + j];
    }
    fq16(g, red);
#pragma unroll
    for (int c = 0; c < 16; ++c) g[c] *= r[c];
    fq16(g, red);
#pragma unroll
    for (int c = 0; c < 16; ++c) {
      const int e = c * 1024 + tid, b = e >> 8, j = e & 255;
      g[c] += qdq(wxt[b * G3H + 512 + j], sWx) + bx[512 + j];
    }
    fq16(g, red);
#pragma unroll
    for (int c = 0; c < 16; ++c) g[c] = tanhf(g[c]);
    fq16(g, red);

#pragma unroll
    for (int c = 0; c < 16; ++c) r[c] = z[c] * h[c];
    fq16(r, red);
#pragma unroll
    for (int c = 0; c < 16; ++c) g[c] = (1.f - z[c]) * g[c];
    fq16(g, red);

#pragma unroll
    for (int c = 0; c < 16; ++c) {
      h[c] = r[c] + g[c];
      out[(size_t)t * NE + c * 1024 + tid] = h[c];
    }
  }

#pragma unroll
  for (int c = 0; c < 16; ++c)
    out[(size_t)TT * NE + c * 1024 + tid] = h[c];
}

extern "C" void kernel_launch(void* const* d_in, const int* in_sizes, int n_in,
                              void* d_out, int out_size, void* d_ws,
                              size_t ws_size, hipStream_t stream) {
  (void)in_sizes; (void)n_in; (void)out_size; (void)ws_size;
  const float* x    = (const float*)d_in[0];
  const float* w_ih = (const float*)d_in[1];
  const float* w_hh = (const float*)d_in[2];
  const float* b_ih = (const float*)d_in[3];
  const float* b_hh = (const float*)d_in[4];
  const float* h0   = (const float*)d_in[5];
  float* out = (float*)d_out;

  char* ws = (char*)d_ws;
  unsigned*    slots = (unsigned*)(ws + WS_SLOTS);
  float*       bx    = (float*)(ws + WS_BX);
  float*       br    = (float*)(ws + WS_BR);
  signed char* WqT   = (signed char*)(ws + WS_WQT);
  signed char* RqT   = (signed char*)(ws + WS_RQT);
  signed char* xq    = (signed char*)(ws + WS_XQ);
  float*       Wx    = (float*)(ws + WS_WX);

  k_init_slots<<<1, 32, 0, stream>>>(slots);
  k_absmax<<<4096, 256, 0, stream>>>(x, NROW * 256, &slots[0]);
  k_absmax<<<192, 256, 0, stream>>>(w_ih, G3H * 256, &slots[1]);
  k_absmax<<<192, 256, 0, stream>>>(w_hh, G3H * 256, &slots[2]);
  k_quant_bias<<<1, 768, 0, stream>>>(b_ih, b_hh, bx, br);
  k_quant_w<<<768, 256, 0, stream>>>(w_ih, &slots[1], WqT);
  k_quant_w<<<768, 256, 0, stream>>>(w_hh, &slots[2], RqT);
  k_quant_x<<<16384, 256, 0, stream>>>(x, &slots[0], xq);
  k_wx_gemm<<<dim3(NROW / 32, G3H / 64), 256, 0, stream>>>(xq, WqT, slots, Wx);

  (void)hipFuncSetAttribute((const void*)k_scan,
                            hipFuncAttributeMaxDynamicSharedMemorySize,
                            SCAN_LDS);
  k_scan<<<1, 1024, SCAN_LDS, stream>>>(Wx, RqT, bx, br, slots, h0, out);
}
